// EDFFN_75453985457390
// MI455X (gfx1250) — matrix operand order repaired and checked by reading, not compiled, not run on hardware
//
#include <hip/hip_runtime.h>
#include <math.h>
#include <stdint.h>

typedef __attribute__((ext_vector_type(16))) _Float16 v16h;
typedef __attribute__((ext_vector_type(8)))  _Float16 v8h;
typedef __attribute__((ext_vector_type(8)))  float    v8f;
typedef __attribute__((ext_vector_type(4)))  float    v4f;
typedef __attribute__((ext_vector_type(4)))  unsigned int u32x4;
typedef __attribute__((ext_vector_type(8)))  int      i32x8;
typedef __attribute__((ext_vector_type(4)))  int      i32x4;

#define DIMC   64
#define HIDC   128
#define C2     256      // 2*HID
#define IMG    256
#define TS     16       // output tile = 16x16 pixels
#define HT     18       // halo tile 18x18
#define HPX    324      // 18*18
#define HPXP   336      // padded to 21 N-tiles of 16
#define XS_STR 72       // f16 stride (64 ic + pad)
#define HS_STR 264      // f16 stride (256 oc + pad)
#define YS_STR 136      // f16 stride (128 c + pad)

struct Smem {
    _Float16 Xs[HPXP * XS_STR];   //  48,384 B : x halo tile, [px][ic] f16
    _Float16 Hs[HPX  * HS_STR];   // 171,072 B : h halo tile, [px][oc] f16 (aliased by Os later)
    _Float16 Ys[256  * YS_STR];   //  69,632 B : GLU output,  [px][c]  f16
    float    Kf[64 * 64];         //  16,384 B : per-channel 8x8 circular-conv kernel
    float    Wd[C2 * 9];          //   9,216 B : depthwise weights (TDM-staged)
};                                // ~314,688 B  (<= 320KB WGP LDS)

__device__ __constant__ float c_cos8[8] = {
    1.0f, 0.70710678118654752f, 0.0f, -0.70710678118654752f,
   -1.0f, -0.70710678118654752f, 0.0f, 0.70710678118654752f };

// A-matrix 16x32 f16 layout: lane L: M=L&15; lanes<16 hold K0..15, lanes>=16 hold K16..31
//                            (contiguous, matching B)
__device__ inline v16h load_fragA_g(const float* W, int ld, int m0, int k0, int lane) {
    const int m  = lane & 15;
    const int hi = lane >> 4;
    const float* p0 = W + (m0 + m) * ld + k0 + hi * 16;
    const float* p1 = p0 + 8;
    v16h a;
#pragma unroll
    for (int i = 0; i < 8; ++i) a[i]     = (_Float16)p0[i];
#pragma unroll
    for (int i = 0; i < 8; ++i) a[8 + i] = (_Float16)p1[i];
    return a;
}

// B-matrix 32x16 f16: lane L: N=L&15; lanes<16 hold K0..15, lanes>=16 hold K16..31 (contiguous)
__device__ inline v16h load_fragB_lds(const _Float16* p) {
    union { v16h v; v8h h[2]; } u;
    u.h[0] = *(const v8h*)p;
    u.h[1] = *(const v8h*)(p + 8);
    return u.v;
}

__device__ inline v8f wmma_f16(v16h a, v16h b, v8f c) {
    return __builtin_amdgcn_wmma_f32_16x16x32_f16(false, a, false, b, (short)0, c, false, false);
}

__device__ inline void load_pair(const _Float16* Hs, int hp, int c, float* d0, float* d1) {
    union { unsigned u; _Float16 h[2]; } t;
    t.u = *(const unsigned*)&Hs[hp * HS_STR + c];
    *d0 = (float)t.h[0]; *d1 = (float)t.h[1];
}

__global__ __launch_bounds__(256) void edffn_fused(
    const float* __restrict__ x,      // (4,64,256,256)
    const float* __restrict__ w_in,   // (256,64)
    const float* __restrict__ w_dw,   // (256,9)
    const float* __restrict__ w_out,  // (64,128)
    const float* __restrict__ fftF,   // (64,8,5)
    float* __restrict__ out)          // (4,64,256,256)
{
    __shared__ Smem S;
    const int tid  = threadIdx.x;
    const int wave = tid >> 5;
    const int lane = tid & 31;

    const int b    = blockIdx.x >> 8;
    const int tile = blockIdx.x & 255;
    const int gy0  = (tile >> 4) * TS;
    const int gx0  = (tile & 15) * TS;

    __builtin_prefetch(w_in,  0, 3);   // global_prefetch_b8: hot small matrices
    __builtin_prefetch(w_out, 0, 3);

    // ---- Stage 0a: depthwise weights -> LDS via Tensor Data Mover (1-D tile, 2304 x f32) ----
#if __has_builtin(__builtin_amdgcn_tensor_load_to_lds) && __has_builtin(__builtin_amdgcn_s_wait_tensorcnt)
    if (wave == 0) {
        const unsigned lds_a = (unsigned)(uintptr_t)&S.Wd[0];          // LDS byte offset
        const unsigned long long ga = (unsigned long long)(uintptr_t)w_dw;
        u32x4 g0;                           // D# group0: count=1, lds_addr, global_addr, type=2
        g0.x = 1u;
        g0.y = lds_a;
        g0.z = (unsigned)ga;
        g0.w = (unsigned)((ga >> 32) & 0x01FFFFFFu) | (2u << 30);
        i32x8 g1;                           // D# group1: data_size=4B, dim0=tile0=2304, dim1=1
        g1.s0 = (int)(2u << 16);
        g1.s1 = (int)(2304u << 16);         // tensor_dim0[15:0] at bits 63:48
        g1.s2 = (int)(1u << 16);            // tensor_dim0[31:16]=0 | tensor_dim1[15:0]=1
        g1.s3 = (int)(2304u << 16);         // tile_dim0 at bits 127:112
        g1.s4 = 0;                          // tile_dim1=0 (1-D), tile_dim2=0
        g1.s5 = 2304;                       // tensor_dim0_stride
        g1.s6 = 0; g1.s7 = 0;
        i32x4 g2 = {0, 0, 0, 0};
        i32x4 g3 = {0, 0, 0, 0};
#if defined(__clang_major__) && __clang_major__ >= 23
        i32x8 gz = {0, 0, 0, 0, 0, 0, 0, 0};
        __builtin_amdgcn_tensor_load_to_lds(g0, g1, g2, g3, gz, 0);
#else
        __builtin_amdgcn_tensor_load_to_lds(g0, g1, g2, g3, 0);
#endif
        __builtin_amdgcn_s_wait_tensorcnt(0);
    }
#else
    for (int idx = tid; idx < C2 * 9; idx += 256) S.Wd[idx] = w_dw[idx];
#endif

    // ---- Stage 0b: x halo tile (f32 -> f16) and circconv kernel K_c = irfft2(F_c) ----
    for (int idx = tid; idx < DIMC * HPXP; idx += 256) {
        const int ic = idx / HPXP;
        const int px = idx % HPXP;
        float v = 0.f;
        if (px < HPX) {
            const int gy = gy0 + (px / HT) - 1;
            const int gx = gx0 + (px % HT) - 1;
            if (gy >= 0 && gy < IMG && gx >= 0 && gx < IMG)
                v = x[((b * DIMC + ic) * IMG + gy) * IMG + gx];
        }
        S.Xs[px * XS_STR + ic] = (_Float16)v;
    }
    // K[u,v] = (1/64) * sum_{q=0..4} lam_q * sum_p F[p,q] cos(2pi(p*u+q*v)/8)
    for (int idx = tid; idx < 64 * 64; idx += 256) {
        const int c = idx >> 6, u = (idx >> 3) & 7, v = idx & 7;
        const float* Fc = fftF + c * 40;
        float s = 0.f;
#pragma unroll
        for (int q = 0; q <= 4; ++q) {
            const float lam = (q == 0 || q == 4) ? 1.f : 2.f;
#pragma unroll
            for (int p = 0; p < 8; ++p)
                s += lam * Fc[p * 5 + q] * c_cos8[(p * u + q * v) & 7];
        }
        S.Kf[idx] = s * (1.f / 64.f);
    }
    __syncthreads();

    // ---- Stage 1: GEMM1 (conv_in): M=256 oc, N=336 px, K=64. Each wave owns mt and mt+8;
    //      B frags read once per nt (shared by both M-tiles), next-B software pipelined ----
    {
        const int mt = wave;
        const v16h a00 = load_fragA_g(w_in, 64, mt * 16,       0,  lane);
        const v16h a01 = load_fragA_g(w_in, 64, mt * 16,       32, lane);
        const v16h a10 = load_fragA_g(w_in, 64, (mt + 8) * 16, 0,  lane);
        const v16h a11 = load_fragA_g(w_in, 64, (mt + 8) * 16, 32, lane);
        const int n   = lane & 15;
        const int hi  = lane >> 4;
        const _Float16* rb0 = &S.Xs[n * XS_STR + hi * 16];
        v16h b0 = load_fragB_lds(rb0);
        v16h b1 = load_fragB_lds(rb0 + 32);
        for (int nt = 0; nt < 21; ++nt) {
            v16h nb0, nb1;
            if (nt < 20) {                         // prefetch next B fragment pair
                const _Float16* rb = rb0 + (nt + 1) * 16 * XS_STR;
                nb0 = load_fragB_lds(rb);
                nb1 = load_fragB_lds(rb + 32);
            }
            v8f acc0 = {}, acc1 = {};
            acc0 = wmma_f16(a00, b0, acc0);
            acc1 = wmma_f16(a10, b0, acc1);
            acc0 = wmma_f16(a01, b1, acc0);
            acc1 = wmma_f16(a11, b1, acc1);
            const int px = nt * 16 + n;
            if (px < HPX) {
                _Float16* hrow = &S.Hs[px * HS_STR + hi * 8];
#pragma unroll
                for (int r = 0; r < 8; ++r) hrow[mt * 16 + r]       = (_Float16)acc0[r];
#pragma unroll
                for (int r = 0; r < 8; ++r) hrow[(mt + 8) * 16 + r] = (_Float16)acc1[r];
            }
            b0 = nb0; b1 = nb1;
        }
    }
    __syncthreads();

    // ---- Stage 2: depthwise 3x3 + exact-GELU GLU. Channel-pair b32 loads + rolling 3-row
    //      window over the column (all indices static after full unroll) ----
    for (int slot = tid; slot < 1024; slot += 256) {
        const int c = (slot >> 4) * 2;            // channel pair c, c+1  (0..126)
        const int j = slot & 15;
        float wA[2][9], wB[2][9];
#pragma unroll
        for (int ch = 0; ch < 2; ++ch)
#pragma unroll
            for (int t = 0; t < 9; ++t) {
                wA[ch][t] = S.Wd[(c + ch) * 9 + t];
                wB[ch][t] = S.Wd[(c + ch + HIDC) * 9 + t];
            }
        float rA[3][3][2], rB[3][3][2];
#pragma unroll
        for (int r = 0; r < 2; ++r)
#pragma unroll
            for (int dj = 0; dj < 3; ++dj) {
                load_pair(S.Hs, r * HT + j + dj, c,        &rA[r][dj][0], &rA[r][dj][1]);
                load_pair(S.Hs, r * HT + j + dj, c + HIDC, &rB[r][dj][0], &rB[r][dj][1]);
            }
#pragma unroll
        for (int i = 0; i < 16; ++i) {
            const int rn = (i + 2) % 3;
#pragma unroll
            for (int dj = 0; dj < 3; ++dj) {
                load_pair(S.Hs, (i + 2) * HT + j + dj, c,        &rA[rn][dj][0], &rA[rn][dj][1]);
                load_pair(S.Hs, (i + 2) * HT + j + dj, c + HIDC, &rB[rn][dj][0], &rB[rn][dj][1]);
            }
            float s1[2] = {0.f, 0.f}, s2[2] = {0.f, 0.f};
#pragma unroll
            for (int di = 0; di < 3; ++di) {
                const int rr = (i + di) % 3;
#pragma unroll
                for (int dj = 0; dj < 3; ++dj)
#pragma unroll
                    for (int ch = 0; ch < 2; ++ch) {
                        s1[ch] += wA[ch][di * 3 + dj] * rA[rr][dj][ch];
                        s2[ch] += wB[ch][di * 3 + dj] * rB[rr][dj][ch];
                    }
            }
            union { unsigned u; _Float16 h[2]; } o;
#pragma unroll
            for (int ch = 0; ch < 2; ++ch) {
                const float g = 0.5f * s1[ch] * (1.0f + erff(s1[ch] * 0.70710678118654752f));
                o.h[ch] = (_Float16)(g * s2[ch]);
            }
            *(unsigned*)&S.Ys[(i * 16 + j) * YS_STR + c] = o.u;
        }
    }
    __syncthreads();

    // ---- Stage 3: GEMM2 (conv_out): M=64, N=256, K=128. D -> Os (f32, aliases Hs) ----
    float* Os = (float*)S.Hs;
    {
        const int mt  = wave & 3;
        const int nt0 = (wave >> 2) * 8;
        v16h a[4];
#pragma unroll
        for (int s = 0; s < 4; ++s) a[s] = load_fragA_g(w_out, 128, mt * 16, s * 32, lane);
        for (int nt = nt0; nt < nt0 + 8; ++nt) {
            v8f acc = {};
            const _Float16* rb = &S.Ys[(nt * 16 + (lane & 15)) * YS_STR + (lane >> 4) * 16];
#pragma unroll
            for (int s = 0; s < 4; ++s)
                acc = wmma_f16(a[s], load_fragB_lds(rb + s * 32), acc);
            const int ocb = mt * 16 + (lane >> 4) * 8;
            const int n   = nt * 16 + (lane & 15);
#pragma unroll
            for (int r = 0; r < 8; ++r)
                Os[(ocb + r) * 256 + n] = acc[r];
        }
    }
    __syncthreads();

    // ---- Stage 4: per-8x8-patch circular convolution. Thread = one 8-wide output row;
    //      o/K rows via b128 vector loads, static circular indices, 8 indep accumulators ----
    for (int rid = tid; rid < 2048; rid += 256) {
        const int oc = rid >> 5;
        const int r5 = rid & 31;
        const int pi = r5 >> 4, pj = (r5 >> 3) & 1, m = r5 & 7;
        const float* obase = &Os[oc * 256 + pj * 8];
        const float* kbase = &S.Kf[oc * 64];
        float acc[8];
#pragma unroll
        for (int n = 0; n < 8; ++n) acc[n] = 0.f;
#pragma unroll
        for (int a = 0; a < 8; ++a) {
            const int ra = (m - a) & 7;
            const float* orow = obase + (pi * 8 + ra) * 16;
            union { v4f v[2]; float f[8]; } ov, kv;
            ov.v[0] = *(const v4f*)(orow);
            ov.v[1] = *(const v4f*)(orow + 4);
            kv.v[0] = *(const v4f*)(kbase + a * 8);
            kv.v[1] = *(const v4f*)(kbase + a * 8 + 4);
#pragma unroll
            for (int n = 0; n < 8; ++n)
#pragma unroll
                for (int bb = 0; bb < 8; ++bb)
                    acc[n] += kv.f[bb] * ov.f[(n - bb) & 7];
        }
        const int gy = gy0 + pi * 8 + m;
        float* op = &out[((b * DIMC + oc) * IMG + gy) * IMG + gx0 + pj * 8];
        union { v4f v[2]; float f[8]; } res;
#pragma unroll
        for (int n = 0; n < 8; ++n) res.f[n] = acc[n];
        *(v4f*)op       = res.v[0];
        *(v4f*)(op + 4) = res.v[1];
    }
}

extern "C" void kernel_launch(void* const* d_in, const int* in_sizes, int n_in,
                              void* d_out, int out_size, void* d_ws, size_t ws_size,
                              hipStream_t stream) {
    (void)in_sizes; (void)n_in; (void)out_size; (void)d_ws; (void)ws_size;
    const float* x     = (const float*)d_in[0];
    const float* w_in  = (const float*)d_in[1];
    const float* w_dw  = (const float*)d_in[2];
    const float* w_out = (const float*)d_in[3];
    const float* fftF  = (const float*)d_in[4];
    float* out = (float*)d_out;
    edffn_fused<<<dim3(4 * 16 * 16), dim3(256), 0, stream>>>(x, w_in, w_dw, w_out, fftF, out);
}
